// SelfAttention_63419487092903
// MI455X (gfx1250) — compile-verified, matrix operand order checked by reading, not run on hardware
//
#include <hip/hip_runtime.h>
#include <hip/hip_bf16.h>

typedef __attribute__((ext_vector_type(16))) __bf16 v16bf;
typedef __attribute__((ext_vector_type(8)))  __bf16 v8bf;
typedef __attribute__((ext_vector_type(4)))  __bf16 v4bf;
typedef __attribute__((ext_vector_type(8)))  float  v8f;

#define EMBED 1024
#define HEADS 16
#define HDIM  64
#define BATCH 4
#define SEQ   2048
#define MROWS (BATCH*SEQ)   // 8192
#define PLD   72            // LDS row stride (elems) for P tile: 144B, 16B-aligned

// ---------------------------------------------------------------------------
// A/B fragment loader for v_wmma_f32_16x16x32_bf16 (NT layout).
// 16-bit A 16x32: lanes 0-15 hold row M=lane, K=0..7 (VGPR0-3), K=16..23
// (VGPR4-7); lanes 16-31 hold K=8..15 / K=24..31. B 32x16 mirrors this with
// lane = column N. Both are two b128 loads from row-major K-contiguous data.
// ---------------------------------------------------------------------------
static __device__ inline v16bf load_frag(const __bf16* base, int ld, int lane, int koff) {
  const int row = lane & 15;
  const int kb  = koff + ((lane >> 4) << 3);
  const __bf16* p = base + row * ld + kb;
  union { v16bf v; v8bf h[2]; } u;
  u.h[0] = *(const v8bf*)(p);
  u.h[1] = *(const v8bf*)(p + 16);
  return u.v;
}

static __device__ inline v16bf lds_frag(const __bf16* base, int lane, int koff) {
  const int row = lane & 15;
  const int kb  = koff + ((lane >> 4) << 3);
  union { v16bf v; v8bf h[2]; } u;
  u.h[0] = *(const v8bf*)(base + row * PLD + kb);
  u.h[1] = *(const v8bf*)(base + row * PLD + kb + 16);
  return u.v;
}

static __device__ inline v8f wmma_bf16(v16bf a, v16bf b, v8f c) {
  return __builtin_amdgcn_wmma_f32_16x16x32_bf16(false, a, false, b, (short)0, c, false, false);
}

// ---------------------------------------------------------------------------
// fp32 -> bf16 conversion
// ---------------------------------------------------------------------------
__global__ __launch_bounds__(256)
void cvt_f32_to_bf16(const float* __restrict__ src, __bf16* __restrict__ dst, int n4) {
  const int i = blockIdx.x * blockDim.x + threadIdx.x;
  if (i < n4) {
    const float4 v = ((const float4*)src)[i];
    v4bf o = { (__bf16)v.x, (__bf16)v.y, (__bf16)v.z, (__bf16)v.w };
    ((v4bf*)dst)[i] = o;
  }
}

// ---------------------------------------------------------------------------
// C[M,N] = A[M,K] . W[N,K]^T  (M=8192, N=K=1024, bf16 in, f32 accum)
// 8 waves/block as 2(M)x4(N): macro tile 128x256; wave tile 64x64 =
// 16 WMMAs per K-chunk of 32 against 8 fragment loads (16x b128).
// MODE 0: bf16 Q/K [B,H,S,D]; MODE 1: bf16 V^T [B,H,D,S]; MODE 2: f32 + bias.
// ---------------------------------------------------------------------------
template<int MODE>
__global__ __launch_bounds__(256)
void gemm_nt(const __bf16* __restrict__ A, const __bf16* __restrict__ W,
             void* __restrict__ Cout, const float* __restrict__ bias) {
  const int lane = threadIdx.x & 31;
  const int wave = threadIdx.x >> 5;
  const int row0 = blockIdx.x * 128 + (wave & 1) * 64;
  const int col0 = blockIdx.y * 256 + (wave >> 1) * 64;

  v8f acc[4][4] = {};
  for (int kk = 0; kk < EMBED; kk += 32) {
    if (kk + 32 < EMBED) {   // hint next K-slab into near caches (global_prefetch_b8)
      __builtin_prefetch(A + (row0 + lane) * EMBED + kk + 32, 0, 3);
      __builtin_prefetch(W + (col0 + lane) * EMBED + kk + 32, 0, 3);
    }
    v16bf af[4], bfr[4];
#pragma unroll
    for (int mt = 0; mt < 4; ++mt)
      af[mt] = load_frag(A + (row0 + mt * 16) * EMBED, EMBED, lane, kk);
#pragma unroll
    for (int nt = 0; nt < 4; ++nt)
      bfr[nt] = load_frag(W + (col0 + nt * 16) * EMBED, EMBED, lane, kk);
#pragma unroll
    for (int mt = 0; mt < 4; ++mt)
#pragma unroll
      for (int nt = 0; nt < 4; ++nt)
        acc[mt][nt] = wmma_bf16(af[mt], bfr[nt], acc[mt][nt]);
  }

  const int mhalf = (lane >> 4) << 3;
  const int ncol  = lane & 15;
#pragma unroll
  for (int mt = 0; mt < 4; ++mt) {
#pragma unroll
    for (int nt = 0; nt < 4; ++nt) {
#pragma unroll
      for (int r = 0; r < 8; ++r) {
        const int m = row0 + mt * 16 + mhalf + r;
        const int n = col0 + nt * 16 + ncol;
        const float v = acc[mt][nt][r];
        if (MODE == 2) {
          ((float*)Cout)[m * EMBED + n] = v + bias[n];
        } else {
          const int b = m >> 11, s = m & (SEQ - 1);
          const int h = n >> 6,  d = n & (HDIM - 1);
          if (MODE == 0)
            ((__bf16*)Cout)[(((b * HEADS + h) * SEQ) + s) * HDIM + d] = (__bf16)v;
          else
            ((__bf16*)Cout)[(((b * HEADS + h) * HDIM) + d) * SEQ + s] = (__bf16)v;
        }
      }
    }
  }
}

// ---------------------------------------------------------------------------
// Causal flash attention: one wave per (b,h,16-query tile), K-chunks of 64.
// Scores: 4x 16x16 tiles (8 WMMAs); online softmax amortizes butterflies and
// O-rescale over 64 columns; P (16x64) transposed via per-wave LDS slice;
// PV: 8 WMMAs (4 d-tiles x 2 K-halves) against V^T rows.
// ---------------------------------------------------------------------------
__global__ __launch_bounds__(128)
void attn_causal(const __bf16* __restrict__ Q, const __bf16* __restrict__ K,
                 const __bf16* __restrict__ Vt, __bf16* __restrict__ AO) {
  __shared__ __bf16 lds[4][16 * PLD];         // 4 waves x 2304B
  const int lane = threadIdx.x & 31;
  const int wave = threadIdx.x >> 5;
  const int task = blockIdx.x * 4 + wave;     // 8192 tasks
  const int qt = task & 127;
  const int bh = task >> 7;
  const int q0 = qt * 16;

  const __bf16* Qp = Q  + (bh * SEQ + q0) * HDIM;
  const __bf16* Kp = K  + bh * SEQ * HDIM;
  const __bf16* Vp = Vt + bh * HDIM * SEQ;
  __bf16* plds = &lds[wave][0];

  const v16bf qf0 = load_frag(Qp, HDIM, lane, 0);
  const v16bf qf1 = load_frag(Qp, HDIM, lane, 32);

  v8f o0 = {}, o1 = {}, o2 = {}, o3 = {};
  float mrow[8], lrow[8];
#pragma unroll
  for (int r = 0; r < 8; ++r) { mrow[r] = -3.0e38f; lrow[r] = 0.0f; }

  const float sc = 0.18033688011112042f;      // (1/sqrt(64)) * log2(e): exp2-domain softmax
  const int mhalf = (lane >> 4) << 3;
  const int ncol  = lane & 15;
  const int nchunk = (q0 + 16 + 63) >> 6;     // causal: only chunks reaching the diagonal

  for (int kc = 0; kc < nchunk; ++kc) {
    const int k0 = kc * 64;
    v8f s[4];
#pragma unroll
    for (int g = 0; g < 4; ++g) {
      const v16bf kf0 = load_frag(Kp + (k0 + g * 16) * HDIM, HDIM, lane, 0);
      const v16bf kf1 = load_frag(Kp + (k0 + g * 16) * HDIM, HDIM, lane, 32);
      v8f z = {};
      z = wmma_bf16(qf0, kf0, z);
      s[g] = wmma_bf16(qf1, kf1, z);
    }

    const bool edge = (kc == nchunk - 1);     // only the diagonal chunk needs masking
#pragma unroll
    for (int r = 0; r < 8; ++r) {
      float x[4];
#pragma unroll
      for (int g = 0; g < 4; ++g) {
        x[g] = s[g][r] * sc;
        if (edge) {
          const int qg = q0 + mhalf + r;
          if (k0 + g * 16 + ncol > qg) x[g] = -1.0e30f;
        }
      }
      float v = fmaxf(fmaxf(x[0], x[1]), fmaxf(x[2], x[3]));
      v = fmaxf(v, __shfl_xor(v, 1));
      v = fmaxf(v, __shfl_xor(v, 2));
      v = fmaxf(v, __shfl_xor(v, 4));
      v = fmaxf(v, __shfl_xor(v, 8));
      const float mnew  = fmaxf(mrow[r], v);
      const float alpha = exp2f(mrow[r] - mnew);
      float p[4], rs = 0.0f;
#pragma unroll
      for (int g = 0; g < 4; ++g) { p[g] = exp2f(x[g] - mnew); rs += p[g]; }
      rs += __shfl_xor(rs, 1);
      rs += __shfl_xor(rs, 2);
      rs += __shfl_xor(rs, 4);
      rs += __shfl_xor(rs, 8);
      lrow[r] = lrow[r] * alpha + rs;
      mrow[r] = mnew;
      o0[r] *= alpha; o1[r] *= alpha; o2[r] *= alpha; o3[r] *= alpha;
      const int m = mhalf + r;
#pragma unroll
      for (int g = 0; g < 4; ++g)
        plds[m * PLD + g * 16 + ncol] = (__bf16)p[g];
    }
    asm volatile("s_wait_dscnt 0x0" ::: "memory");
    const v16bf pf0 = lds_frag(plds, lane, 0);
    const v16bf pf1 = lds_frag(plds, lane, 32);

    o0 = wmma_bf16(pf0, load_frag(Vp +  0 * SEQ, SEQ, lane, k0     ), o0);
    o0 = wmma_bf16(pf1, load_frag(Vp +  0 * SEQ, SEQ, lane, k0 + 32), o0);
    o1 = wmma_bf16(pf0, load_frag(Vp + 16 * SEQ, SEQ, lane, k0     ), o1);
    o1 = wmma_bf16(pf1, load_frag(Vp + 16 * SEQ, SEQ, lane, k0 + 32), o1);
    o2 = wmma_bf16(pf0, load_frag(Vp + 32 * SEQ, SEQ, lane, k0     ), o2);
    o2 = wmma_bf16(pf1, load_frag(Vp + 32 * SEQ, SEQ, lane, k0 + 32), o2);
    o3 = wmma_bf16(pf0, load_frag(Vp + 48 * SEQ, SEQ, lane, k0     ), o3);
    o3 = wmma_bf16(pf1, load_frag(Vp + 48 * SEQ, SEQ, lane, k0 + 32), o3);
  }

  const int b = bh >> 4;
  const int h = bh & 15;
#pragma unroll
  for (int r = 0; r < 8; ++r) {
    const float inv = 1.0f / lrow[r];
    const int s = q0 + mhalf + r;
    __bf16* dst = AO + (b * SEQ + s) * EMBED + h * HDIM + ncol;
    dst[ 0] = (__bf16)(o0[r] * inv);
    dst[16] = (__bf16)(o1[r] * inv);
    dst[32] = (__bf16)(o2[r] * inv);
    dst[48] = (__bf16)(o3[r] * inv);
  }
}

// ---------------------------------------------------------------------------
extern "C" void kernel_launch(void* const* d_in, const int* in_sizes, int n_in,
                              void* d_out, int out_size, void* d_ws, size_t ws_size,
                              hipStream_t stream) {
  const float* x  = (const float*)d_in[0];
  const float* Wq = (const float*)d_in[1];
  const float* Wk = (const float*)d_in[2];
  const float* Wv = (const float*)d_in[3];
  const float* Wo = (const float*)d_in[4];
  const float* bo = (const float*)d_in[5];

  char* ws = (char*)d_ws;
  const size_t SZ_X = (size_t)MROWS * EMBED * 2;   // 16 MB
  const size_t SZ_W = (size_t)EMBED * EMBED * 2;   //  2 MB
  __bf16* xh  = (__bf16*)(ws);
  __bf16* wqh = (__bf16*)(ws + SZ_X);
  __bf16* wkh = (__bf16*)(ws + SZ_X + 1 * SZ_W);
  __bf16* wvh = (__bf16*)(ws + SZ_X + 2 * SZ_W);
  __bf16* woh = (__bf16*)(ws + SZ_X + 3 * SZ_W);
  __bf16* qh  = (__bf16*)(ws + 1 * SZ_X + 4 * SZ_W);
  __bf16* kh  = (__bf16*)(ws + 2 * SZ_X + 4 * SZ_W);
  __bf16* vth = (__bf16*)(ws + 3 * SZ_X + 4 * SZ_W);
  __bf16* aoh = (__bf16*)(ws + 4 * SZ_X + 4 * SZ_W);  // total ~88 MB

  const int nx4 = MROWS * EMBED / 4;
  const int nw4 = EMBED * EMBED / 4;
  cvt_f32_to_bf16<<<(nx4 + 255) / 256, 256, 0, stream>>>(x,  xh,  nx4);
  cvt_f32_to_bf16<<<(nw4 + 255) / 256, 256, 0, stream>>>(Wq, wqh, nw4);
  cvt_f32_to_bf16<<<(nw4 + 255) / 256, 256, 0, stream>>>(Wk, wkh, nw4);
  cvt_f32_to_bf16<<<(nw4 + 255) / 256, 256, 0, stream>>>(Wv, wvh, nw4);
  cvt_f32_to_bf16<<<(nw4 + 255) / 256, 256, 0, stream>>>(Wo, woh, nw4);

  dim3 ggrid(MROWS / 128, EMBED / 256);   // 64 x 4 workgroups, 8 waves each
  gemm_nt<0><<<ggrid, 256, 0, stream>>>(xh, wqh, qh,  nullptr);
  gemm_nt<0><<<ggrid, 256, 0, stream>>>(xh, wkh, kh,  nullptr);
  gemm_nt<1><<<ggrid, 256, 0, stream>>>(xh, wvh, vth, nullptr);

  attn_causal<<<(BATCH * HEADS * (SEQ / 16)) / 4, 128, 0, stream>>>(qh, kh, vth, aoh);

  gemm_nt<2><<<ggrid, 256, 0, stream>>>(aoh, woh, d_out, bo);
}
